// Net_vanilla_34591666602131
// MI455X (gfx1250) — compile-verified, matrix operand order checked by reading, not run on hardware
//
#include <hip/hip_runtime.h>
#include <hip/hip_bf16.h>
#include <math.h>

typedef __attribute__((ext_vector_type(16))) _Float16 v16h;
typedef __attribute__((ext_vector_type(8)))  float    v8f;

#define GN_N     50000
#define GN_E     800000
#define GN_FIN   26
#define GN_H     64
#define GN_L     6
#define GN_TILES 3125   // 50000 / 16, exact

// ---------------------------------------------------------------------------
// WMMA helpers: 16x16x32 f16 -> f32, wave32.
// A 16x32 (f16): lane L holds row m=L%16; VGPR v holds K pair
//   k = kc*32 + (v>=4?16:0) + (L>=16?8:0) + 2*(v&3) + bit
// B 32x16 (f16): symmetric, lane L holds column n=L%16, same K pattern.
// C/D 16x16 (f32): lane L col n=L%16, VGPR v row m = v + (L>=16?8:0).
// ---------------------------------------------------------------------------
__device__ __forceinline__ v8f wmma_f16(v16h a, v16h b, v8f c) {
  return __builtin_amdgcn_wmma_f32_16x16x32_f16(false, a, false, b, (short)0, c,
                                                false, false);
}

// Build an A fragment (K chunk kc in {0,1}) from an f16 row-major [*, 64] buffer.
__device__ __forceinline__ v16h load_a_f16(const _Float16* __restrict__ buf,
                                           int rowbase, int lane, int kc) {
  const _Float16* row = buf + (size_t)(rowbase + (lane & 15)) * GN_H;
  const int koff = kc * 32 + ((lane >> 4) & 1) * 8;
  v16h a;
#pragma unroll
  for (int v = 0; v < 8; ++v) {
    const int k = koff + ((v >= 4) ? 16 : 0) + 2 * (v & 3);
    a[2 * v]     = row[k];
    a[2 * v + 1] = row[k + 1];
  }
  return a;
}

// Build an A fragment from an f32 row-major [*, 64] buffer (convert to f16).
__device__ __forceinline__ v16h load_a_f32(const float* __restrict__ buf,
                                           int rowbase, int lane, int kc) {
  const float* row = buf + (size_t)(rowbase + (lane & 15)) * GN_H;
  const int koff = kc * 32 + ((lane >> 4) & 1) * 8;
  v16h a;
#pragma unroll
  for (int v = 0; v < 8; ++v) {
    const int k = koff + ((v >= 4) ? 16 : 0) + 2 * (v & 3);
    const float2 f = *(const float2*)(row + k);  // k even -> 8B aligned
    a[2 * v]     = (_Float16)f.x;
    a[2 * v + 1] = (_Float16)f.y;
  }
  return a;
}

// Pre-swizzled B fragment: contiguous 32B per lane, coalesced across the wave.
__device__ __forceinline__ v16h load_b_sw(const _Float16* __restrict__ sw,
                                          int tile, int kc, int lane) {
  return *(const v16h*)(sw + ((size_t)(tile * 2 + kc) * 32 + lane) * 16);
}

__device__ __forceinline__ float sigf(float x) {
  return 1.0f / (1.0f + expf(-x));
}

// ---------------------------------------------------------------------------
// Weight swizzle: place f16 weights in exact B-fragment layout.
//  Wih_sw / Whh_sw : 12 tiles (N=192) x 2 kc  -> B[k][n] = W[n*64 + k]  (W.T)
//  Wconv_sw        : 6 layers x 4 tiles x 2 kc-> B[k][n] = conv_W[(l*64+k)*64+n]
// ---------------------------------------------------------------------------
__global__ void prep_weights_kernel(const float* __restrict__ W_ih,
                                    const float* __restrict__ W_hh,
                                    const float* __restrict__ conv_W,
                                    _Float16* __restrict__ Wih_sw,
                                    _Float16* __restrict__ Whh_sw,
                                    _Float16* __restrict__ Wconv_sw) {
  const int tid = blockIdx.x * blockDim.x + threadIdx.x;
  if (tid >= 49152) return;
  if (tid < 24576) {
    const float* src = (tid < 12288) ? W_ih : W_hh;
    _Float16*    dst = (tid < 12288) ? Wih_sw : Whh_sw;
    const int idx  = tid & 12287;
    const int j    = idx & 15;
    const int lane = (idx >> 4) & 31;
    const int kc   = (idx >> 9) & 1;
    const int t    = idx >> 10;  // 0..11
    const int v    = j >> 1;
    const int k    = kc * 32 + ((v >= 4) ? 16 : 0) + ((lane >= 16) ? 8 : 0) +
                     2 * (v & 3) + (j & 1);
    const int n    = t * 16 + (lane & 15);        // 0..191
    dst[idx] = (_Float16)src[n * GN_H + k];       // transpose: B[k][n] = W[n][k]
  } else {
    const int idx2 = tid - 24576;
    const int l    = idx2 >> 12;  // layer
    const int r    = idx2 & 4095;
    const int j    = r & 15;
    const int lane = (r >> 4) & 31;
    const int kc   = (r >> 9) & 1;
    const int t    = r >> 10;     // 0..3
    const int v    = j >> 1;
    const int k    = kc * 32 + ((v >= 4) ? 16 : 0) + ((lane >= 16) ? 8 : 0) +
                     2 * (v & 3) + (j & 1);
    const int n    = t * 16 + (lane & 15);
    Wconv_sw[idx2] = (_Float16)conv_W[(size_t)(l * GN_H + k) * GN_H + n];
  }
}

// ---------------------------------------------------------------------------
// lin0: h = sigmoid(x @ lin0_W), also emits x1 output and f16 copy of h.
// K=26 is tiny -> VALU kernel (~1% of total FLOPs).
// ---------------------------------------------------------------------------
__global__ void lin0_kernel(const float* __restrict__ x,
                            const float* __restrict__ W,
                            float* __restrict__ h32,
                            _Float16* __restrict__ h16,
                            float* __restrict__ x1_out) {
  const int tid = blockIdx.x * blockDim.x + threadIdx.x;
  if (tid >= GN_N * GN_H) return;
  const int nrow = tid >> 6;
  const int c    = tid & 63;
  const float* xr = x + (size_t)nrow * GN_FIN;
  float s = 0.0f;
#pragma unroll
  for (int k = 0; k < GN_FIN; ++k) s += xr[k] * W[k * GN_H + c];
  const float val = sigf(s);
  h32[tid]    = val;
  h16[tid]    = (_Float16)val;
  x1_out[tid] = val;
}

__global__ void zero_kernel(float4* __restrict__ p, int n4) {
  const int tid = blockIdx.x * blockDim.x + threadIdx.x;
  if (tid < n4) p[tid] = make_float4(0.f, 0.f, 0.f, 0.f);
}

// ---------------------------------------------------------------------------
// conv: m = h @ conv_W[l]   (N x 64) via WMMA, one wave per 16-row tile.
// ---------------------------------------------------------------------------
__global__ void conv_kernel(const _Float16* __restrict__ h16,
                            const _Float16* __restrict__ Bsw,  // this layer, 4096 halves
                            float* __restrict__ m32) {
  const int wave = blockIdx.x * (blockDim.x >> 5) + (threadIdx.x >> 5);
  if (wave >= GN_TILES) return;  // wave-uniform: EXEC stays all-ones for WMMA
  const int lane = threadIdx.x & 31;
  const int base = wave * 16;

  v8f acc[4];
#pragma unroll
  for (int t = 0; t < 4; ++t) { v8f z = {}; acc[t] = z; }

#pragma unroll
  for (int kc = 0; kc < 2; ++kc) {
    const v16h a = load_a_f16(h16, base, lane, kc);
#pragma unroll
    for (int t = 0; t < 4; ++t)
      acc[t] = wmma_f16(a, load_b_sw(Bsw, t, kc, lane), acc[t]);
  }

  const int col0 = lane & 15;
  const int moff = ((lane >> 4) & 1) * 8;
#pragma unroll
  for (int t = 0; t < 4; ++t)
#pragma unroll
    for (int v = 0; v < 8; ++v)
      m32[(size_t)(base + v + moff) * GN_H + t * 16 + col0] = acc[t][v];
}

// ---------------------------------------------------------------------------
// scatter: agg[dst[e]] += m[src[e]]  (f32 HW atomics, L2-resident buffers).
// 16 threads per edge, float4 each (64 floats/edge).
// ---------------------------------------------------------------------------
__global__ void scatter_kernel(const float* __restrict__ m32,
                               const long long* __restrict__ edges,  // [2,E] int64
                               float* __restrict__ agg) {
  const int tid = blockIdx.x * blockDim.x + threadIdx.x;
  if (tid >= GN_E * 16) return;
  const int e    = tid >> 4;
  const int part = tid & 15;
  const long long s = edges[e];
  const long long d = edges[GN_E + e];
  const float4 v = *(const float4*)(m32 + (size_t)s * GN_H + part * 4);
  float* dp = agg + (size_t)d * GN_H + part * 4;
  __hip_atomic_fetch_add(dp + 0, v.x, __ATOMIC_RELAXED, __HIP_MEMORY_SCOPE_AGENT);
  __hip_atomic_fetch_add(dp + 1, v.y, __ATOMIC_RELAXED, __HIP_MEMORY_SCOPE_AGENT);
  __hip_atomic_fetch_add(dp + 2, v.z, __ATOMIC_RELAXED, __HIP_MEMORY_SCOPE_AGENT);
  __hip_atomic_fetch_add(dp + 3, v.w, __ATOMIC_RELAXED, __HIP_MEMORY_SCOPE_AGENT);
}

// ---------------------------------------------------------------------------
// Fused GRU: gi = agg@Wih.T, gh = h@Whh.T, gates, h' writeback (f32 + f16).
// r/z blocks accumulate BOTH products into the same WMMA accumulator
// (i_r+h_r / i_z+h_z come out for free); n block kept split for r*h_n.
// 48 WMMAs per 16-node tile. One wave per tile.
// ---------------------------------------------------------------------------
__global__ void gru_kernel(const float* __restrict__ agg,
                           _Float16* __restrict__ h16,
                           float* __restrict__ h32,
                           const _Float16* __restrict__ Bih,
                           const _Float16* __restrict__ Bhh,
                           const float* __restrict__ b_ih,
                           const float* __restrict__ b_hh) {
  const int wave = blockIdx.x * (blockDim.x >> 5) + (threadIdx.x >> 5);
  if (wave >= GN_TILES) return;
  const int lane = threadIdx.x & 31;
  const int base = wave * 16;

  v8f rz[8], in_[4], hn[4];
#pragma unroll
  for (int t = 0; t < 8; ++t) { v8f z = {}; rz[t] = z; }
#pragma unroll
  for (int t = 0; t < 4; ++t) { v8f z = {}; in_[t] = z; hn[t] = z; }

#pragma unroll
  for (int kc = 0; kc < 2; ++kc) {
    const v16h aA = load_a_f32(agg, base, lane, kc);  // agg tile, cvt f32->f16
    const v16h aH = load_a_f16(h16, base, lane, kc);  // h tile
#pragma unroll
    for (int t = 0; t < 8; ++t) {  // r (tiles 0..3) and z (tiles 4..7) blocks
      rz[t] = wmma_f16(aA, load_b_sw(Bih, t, kc, lane), rz[t]);
      rz[t] = wmma_f16(aH, load_b_sw(Bhh, t, kc, lane), rz[t]);
    }
#pragma unroll
    for (int t = 0; t < 4; ++t) {  // n block (tiles 8..11), kept separate
      in_[t] = wmma_f16(aA, load_b_sw(Bih, 8 + t, kc, lane), in_[t]);
      hn[t]  = wmma_f16(aH, load_b_sw(Bhh, 8 + t, kc, lane), hn[t]);
    }
  }

  const int col0 = lane & 15;
  const int moff = ((lane >> 4) & 1) * 8;
#pragma unroll
  for (int t = 0; t < 4; ++t) {
    const int col = t * 16 + col0;           // output feature column 0..63
    const float br  = b_ih[col]       + b_hh[col];
    const float bz  = b_ih[64 + col]  + b_hh[64 + col];
    const float bin = b_ih[128 + col];
    const float bhn = b_hh[128 + col];
#pragma unroll
    for (int v = 0; v < 8; ++v) {
      const size_t idx = (size_t)(base + v + moff) * GN_H + col;
      const float r  = sigf(rz[t][v] + br);
      const float z  = sigf(rz[t + 4][v] + bz);
      const float n  = tanhf(in_[t][v] + bin + r * (hn[t][v] + bhn));
      const float ho = h32[idx];
      const float hn_new = (1.0f - z) * n + z * ho;
      h32[idx] = hn_new;
      h16[idx] = (_Float16)hn_new;
    }
  }
}

// ---------------------------------------------------------------------------
// lin1: out[n] = relu(h[n]) . lin1_W + b. One wave per node, shuffle-reduce.
// ---------------------------------------------------------------------------
__global__ void lin1_kernel(const float* __restrict__ h32,
                            const float* __restrict__ W,
                            const float* __restrict__ b,
                            float* __restrict__ out) {
  const int wave = blockIdx.x * (blockDim.x >> 5) + (threadIdx.x >> 5);
  if (wave >= GN_N) return;
  const int lane = threadIdx.x & 31;
  const float* hr = h32 + (size_t)wave * GN_H;
  float s = 0.0f;
#pragma unroll
  for (int i = 0; i < 2; ++i) {
    const int c = lane + i * 32;
    float hv = hr[c];
    hv = hv > 0.0f ? hv : 0.0f;
    s += hv * W[c];
  }
#pragma unroll
  for (int off = 16; off > 0; off >>= 1) s += __shfl_xor(s, off, 32);
  if (lane == 0) out[wave] = s + b[0];
}

// ---------------------------------------------------------------------------
// Launch. Workspace layout (all offsets 256B-aligned), ~44.9 MB total:
//   h32[N*64 f32] | h16[N*64 f16] | m32[N*64 f32] | agg[N*64 f32]
//   | Wih_sw[12288 f16] | Whh_sw[12288 f16] | Wconv_sw[24576 f16]
// ---------------------------------------------------------------------------
extern "C" void kernel_launch(void* const* d_in, const int* in_sizes, int n_in,
                              void* d_out, int out_size, void* d_ws, size_t ws_size,
                              hipStream_t stream) {
  const float*     x      = (const float*)d_in[0];
  const long long* edges  = (const long long*)d_in[1];  // int64 [2,E]
  const float*     lin0_W = (const float*)d_in[2];
  const float*     conv_W = (const float*)d_in[3];
  const float*     W_ih   = (const float*)d_in[4];
  const float*     W_hh   = (const float*)d_in[5];
  const float*     b_ih   = (const float*)d_in[6];
  const float*     b_hh   = (const float*)d_in[7];
  const float*     lin1_W = (const float*)d_in[8];
  const float*     lin1_b = (const float*)d_in[9];

  float* out_scalar = (float*)d_out;          // out[:,0], N floats
  float* out_x1     = (float*)d_out + GN_N;   // x1, N*64 floats

  char* ws = (char*)d_ws;
  float*     h32      = (float*)(ws);
  _Float16*  h16      = (_Float16*)(ws + 12800000);
  float*     m32      = (float*)(ws + 19200000);
  float*     agg      = (float*)(ws + 32000000);
  _Float16*  Wih_sw   = (_Float16*)(ws + 44800000);
  _Float16*  Whh_sw   = (_Float16*)(ws + 44824576);
  _Float16*  Wconv_sw = (_Float16*)(ws + 44849152);

  // 1) one-time weight swizzle into WMMA B-fragment layout
  prep_weights_kernel<<<(49152 + 255) / 256, 256, 0, stream>>>(
      W_ih, W_hh, conv_W, Wih_sw, Whh_sw, Wconv_sw);

  // 2) lin0 + sigmoid -> h (f32/f16) and x1 output
  lin0_kernel<<<(GN_N * GN_H + 255) / 256, 256, 0, stream>>>(
      x, lin0_W, h32, h16, out_x1);

  // 3) six GatedGraphConv layers
  const int tile_blocks = (GN_TILES + 3) / 4;  // 128 threads = 4 waves/block
  for (int l = 0; l < GN_L; ++l) {
    zero_kernel<<<(GN_N * GN_H / 4 + 255) / 256, 256, 0, stream>>>(
        (float4*)agg, GN_N * GN_H / 4);
    conv_kernel<<<tile_blocks, 128, 0, stream>>>(
        h16, Wconv_sw + (size_t)l * 4096, m32);
    scatter_kernel<<<(GN_E * 16 + 255) / 256, 256, 0, stream>>>(
        m32, edges, agg);
    gru_kernel<<<tile_blocks, 128, 0, stream>>>(
        agg, h16, h32, Wih_sw, Whh_sw, b_ih, b_hh);
  }

  // 4) readout
  lin1_kernel<<<(GN_N + 7) / 8, 256, 0, stream>>>(h32, lin1_W, lin1_b, out_scalar);
}
